// LayoutLMv3SelfAttention_89532888252843
// MI455X (gfx1250) — matrix operand order repaired and checked by reading, not compiled, not run on hardware
//
#include <hip/hip_runtime.h>
#include <hip/hip_bf16.h>

// ---- types -----------------------------------------------------------------
typedef __bf16 bf16_t;
typedef __attribute__((ext_vector_type(16))) __bf16 v16bf;
typedef __attribute__((ext_vector_type(8)))  __bf16 v8bf;
typedef __attribute__((ext_vector_type(8)))  float  v8f;
typedef __attribute__((ext_vector_type(8)))  float  v8fv;

#define BATCH 4
#define HEADS 16
#define SEQ   1024
#define CDIM  1024
#define DHEAD 64
#define INV_SQRT_D 0.125f

// ============================================================================
// Kernel 1: fused QKV projection.  y = x @ W^T + b.
// One wave computes a 16-row x 64-col tile (a 64-col tile == one head).
// Q is pre-scaled by 1/sqrt(D). V is stored transposed [BH, D, S] so the
// P*V WMMA B-fragment in kernel 2 is a contiguous load.
// ============================================================================
__global__ __launch_bounds__(128) void qkv_proj_kernel(
    const float* __restrict__ x,                       // [4096, 1024]
    const float* __restrict__ Wq, const float* __restrict__ bq,
    const float* __restrict__ Wk, const float* __restrict__ bk,
    const float* __restrict__ Wv, const float* __restrict__ bv,
    bf16_t* __restrict__ Qo,                           // [BH, S, D] (scaled)
    bf16_t* __restrict__ Ko,                           // [BH, S, D]
    bf16_t* __restrict__ Vt)                           // [BH, D, S]
{
  const int w  = threadIdx.x >> 5;      // wave in block: 0..3
  const int l  = threadIdx.x & 31;      // lane
  const int g  = l >> 4;                // half-wave group
  const int ln = l & 15;

  const int mtile = blockIdx.x * 4 + w; // 0..255 -> 16-row tile of [4096 rows]
  const int h     = blockIdx.y;         // 0..15  -> head (= 64-col tile)
  const int which = blockIdx.z;         // 0=Q 1=K 2=V

  const float* W    = (which == 0) ? Wq : (which == 1) ? Wk : Wv;
  const float* bias = (which == 0) ? bq : (which == 1) ? bk : bv;

  const int m0 = mtile * 16;

  v8f acc[4] = {v8f{0.f,0.f,0.f,0.f,0.f,0.f,0.f,0.f},
                v8f{0.f,0.f,0.f,0.f,0.f,0.f,0.f,0.f},
                v8f{0.f,0.f,0.f,0.f,0.f,0.f,0.f,0.f},
                v8f{0.f,0.f,0.f,0.f,0.f,0.f,0.f,0.f}};

  for (int k0 = 0; k0 < CDIM; k0 += 32) {
    // A-fragment (16x32 bf16): lane holds row m0+ln, K chunks [g*8, g*8+8)
    // and [16+g*8, 16+g*8+8)  (ISA 7.12.2, 16-bit A 16x32 layout).
    v16bf a;
    {
      const float* xr = x + (size_t)(m0 + ln) * CDIM + k0 + g * 8;
      v8fv lo = *(const v8fv*)xr;
      v8fv hi = *(const v8fv*)(xr + 16);
      #pragma unroll
      for (int t = 0; t < 8; ++t) { a[t] = (__bf16)lo[t]; a[8 + t] = (__bf16)hi[t]; }
    }
    // Four B-fragments (32x16 bf16): lane holds column n = h*64+ft*16+ln,
    // K chunks [g*8, g*8+8) and [16+g*8, 16+g*8+8).
    #pragma unroll
    for (int ft = 0; ft < 4; ++ft) {
      const int n = h * 64 + ft * 16 + ln;
      const float* wr = W + (size_t)n * CDIM + k0 + g * 8;
      v8fv w0 = *(const v8fv*)wr;
      v8fv w1 = *(const v8fv*)(wr + 16);
      v16bf bfr;
      #pragma unroll
      for (int t = 0; t < 8; ++t) { bfr[t] = (__bf16)w0[t]; bfr[8 + t] = (__bf16)w1[t]; }
      acc[ft] = __builtin_amdgcn_wmma_f32_16x16x32_bf16(
          false, a, false, bfr, (short)0, acc[ft], false, false);
    }
  }

  // Epilogue: bias add, optional scaling, scatter as bf16.
  const int b  = m0 >> 10;            // batch
  const int s0 = m0 & (SEQ - 1);      // row within batch (tiles never cross b)
  const int bh = b * HEADS + h;
  #pragma unroll
  for (int ft = 0; ft < 4; ++ft) {
    const int d = ft * 16 + ln;
    const float bval = bias[h * 64 + d];
    #pragma unroll
    for (int i = 0; i < 8; ++i) {
      const int s = s0 + i + 8 * g;   // C-layout: VGPR i -> row i + 8*g
      const float val = acc[ft][i] + bval;
      if (which == 0)
        Qo[((size_t)bh * SEQ + s) * DHEAD + d] = (__bf16)(val * INV_SQRT_D);
      else if (which == 1)
        Ko[((size_t)bh * SEQ + s) * DHEAD + d] = (__bf16)val;
      else
        Vt[((size_t)bh * DHEAD + d) * SEQ + s] = (__bf16)val;
    }
  }
}

// ============================================================================
// Kernel 2: flash attention. One wave owns a 16-query tile; streams 32 keys
// per iteration. The 512MB rel_pos/rel_2d_pos stream (the HBM bottleneck) is
// double-buffered into LDS via CDNA5 async-to-LDS loads (ASYNCcnt), so the
// score WMMAs of iteration i hide the HBM latency of tile i+1. Async loads
// complete in issue order, so s_wait_asynccnt 8 (= the next tile's 8
// outstanding b128 requests) guarantees the current tile has landed.
// ============================================================================
__global__ __launch_bounds__(128) void flash_attn_kernel(
    const bf16_t* __restrict__ Qs,    // [BH, S, D] pre-scaled
    const bf16_t* __restrict__ Ks,    // [BH, S, D]
    const bf16_t* __restrict__ Vts,   // [BH, D, S]
    const float*  __restrict__ rel,   // [BH, S, S]
    const float*  __restrict__ rel2,  // [BH, S, S]
    const float*  __restrict__ mask,  // [B, S]
    float* __restrict__ out)          // [B, S, C]
{
  __shared__ __align__(16) __bf16 pshare[4][16][32];
  // [wave][buf][tensor][16 rows * 32 cols] f32 bias staging tiles (2KB each)
  __shared__ __align__(16) float relbuf[4][2][2][16 * 32];

  const int w  = threadIdx.x >> 5;
  const int l  = threadIdx.x & 31;
  const int g  = l >> 4;
  const int ln = l & 15;

  const int bh = blockIdx.x;
  const int b  = bh >> 4;
  const int h  = bh & 15;
  const int q0 = (blockIdx.y * 4 + w) * 16;

  const bf16_t* Q = Qs  + (size_t)bh * SEQ * DHEAD;
  const bf16_t* K = Ks  + (size_t)bh * SEQ * DHEAD;
  const bf16_t* V = Vts + (size_t)bh * DHEAD * SEQ;
  const float* relp  = rel  + (size_t)bh * SEQ * SEQ;
  const float* rel2p = rel2 + (size_t)bh * SEQ * SEQ;
  const float* mrow  = mask + (size_t)b * SEQ;

  // Issue one 16x32 f32 tile of each bias tensor into LDS buffer `buf`.
  // Tile = 2KB per tensor = 4 async b128 issues per tensor (32 lanes x 16B).
  auto issue_bias_tile = [&](int kbn, int buf) {
    #pragma unroll
    for (int t = 0; t < 2; ++t) {
      const float* src = (t == 0) ? relp : rel2p;
      const unsigned ldsb =
          (unsigned)(unsigned long long)(const void*)&relbuf[w][buf][t][0];
      #pragma unroll
      for (int inst = 0; inst < 4; ++inst) {
        const int c = inst * 32 + l;               // 16B chunk id: 0..127
        const unsigned long long ga = (unsigned long long)(const void*)
            (src + (size_t)(q0 + (c >> 3)) * SEQ + kbn + (c & 7) * 4);
        const unsigned la = ldsb + (unsigned)c * 16u;
        asm volatile("global_load_async_to_lds_b128 %0, %1, off"
                     :: "v"(la), "v"(ga) : "memory");
      }
    }
  };

  // Q A-fragments (held for the whole key loop)
  v16bf aQ[2];
  #pragma unroll
  for (int dh = 0; dh < 2; ++dh) {
    const bf16_t* qr = Q + (size_t)(q0 + ln) * DHEAD + dh * 32 + g * 16;
    v8bf lo = *(const v8bf*)qr;
    v8bf hi = *(const v8bf*)(qr + 8);
    aQ[dh] = __builtin_shufflevector(lo, hi, 0,1,2,3,4,5,6,7,8,9,10,11,12,13,14,15);
  }

  float mprev[8], lsum[8];
  #pragma unroll
  for (int i = 0; i < 8; ++i) { mprev[i] = -3.0e38f; lsum[i] = 0.f; }
  v8f ctx[4] = {v8f{0.f,0.f,0.f,0.f,0.f,0.f,0.f,0.f},
                v8f{0.f,0.f,0.f,0.f,0.f,0.f,0.f,0.f},
                v8f{0.f,0.f,0.f,0.f,0.f,0.f,0.f,0.f},
                v8f{0.f,0.f,0.f,0.f,0.f,0.f,0.f,0.f}};

  issue_bias_tile(0, 0);   // prologue: stage the first bias tile

  for (int kb = 0; kb < SEQ; kb += 32) {
    const int cur = (kb >> 5) & 1;
    const bool have_next = (kb + 32) < SEQ;
    if (have_next) issue_bias_tile(kb + 32, cur ^ 1);  // prefetch next tile

    // ---- scores: S = Q * K^T  (two 16-key subtiles c, K-dim = D in 2 halves)
    v8f sc[2] = {v8f{0.f,0.f,0.f,0.f,0.f,0.f,0.f,0.f},
                 v8f{0.f,0.f,0.f,0.f,0.f,0.f,0.f,0.f}};
    #pragma unroll
    for (int c = 0; c < 2; ++c) {
      #pragma unroll
      for (int dh = 0; dh < 2; ++dh) {
        const bf16_t* kr = K + (size_t)(kb + c * 16 + ln) * DHEAD + dh * 32 + g * 16;
        v16bf bK = *(const v16bf*)kr;
        sc[c] = __builtin_amdgcn_wmma_f32_16x16x32_bf16(
            false, aQ[dh], false, bK, (short)0, sc[c], false, false);
      }
    }

    // ---- wait for the CURRENT bias tile (next tile's 8 may stay in flight)
    if (have_next) asm volatile("s_wait_asynccnt 8" ::: "memory");
    else           asm volatile("s_wait_asynccnt 0" ::: "memory");
    const float* r0 = &relbuf[w][cur][0][0];
    const float* r1 = &relbuf[w][cur][1][0];

    // ---- fused bias: (rel + rel2) * inv_sqrt_d + attention_mask
    float p0[8], p1[8], mt[8];
    #pragma unroll
    for (int i = 0; i < 8; ++i) {
      const int ro = (i + 8 * g) * 32 + ln;
      p0[i] = sc[0][i] + (r0[ro]      + r1[ro])      * INV_SQRT_D + mrow[kb + ln];
      p1[i] = sc[1][i] + (r0[ro + 16] + r1[ro + 16]) * INV_SQRT_D + mrow[kb + 16 + ln];
      float m = fmaxf(p0[i], p1[i]);
      #pragma unroll
      for (int off = 1; off < 16; off <<= 1)       // stays within 16-lane group
        m = fmaxf(m, __shfl_xor(m, off, 32));
      mt[i] = m;
    }

    // ---- online softmax update + write P tile to LDS (C-layout -> memory)
    #pragma unroll
    for (int i = 0; i < 8; ++i) {
      const float mn    = fmaxf(mprev[i], mt[i]);
      const float scale = __expf(mprev[i] - mn);
      mprev[i] = mn;
      const float e0 = __expf(p0[i] - mn);
      const float e1 = __expf(p1[i] - mn);
      float rs = e0 + e1;
      #pragma unroll
      for (int off = 1; off < 16; off <<= 1)
        rs += __shfl_xor(rs, off, 32);
      lsum[i] = lsum[i] * scale + rs;
      #pragma unroll
      for (int dt = 0; dt < 4; ++dt) ctx[dt][i] *= scale;
      pshare[w][i + 8 * g][ln]      = (__bf16)e0;
      pshare[w][i + 8 * g][16 + ln] = (__bf16)e1;
    }

    // ---- reload P as A-fragment (same wave, DS ops in-order)
    v8bf plo = *(const v8bf*)&pshare[w][ln][g * 16];
    v8bf phi = *(const v8bf*)&pshare[w][ln][g * 16 + 8];
    v16bf aP = __builtin_shufflevector(plo, phi, 0,1,2,3,4,5,6,7,8,9,10,11,12,13,14,15);

    // ---- ctx += P * V   (V transposed: contiguous along keys)
    #pragma unroll
    for (int dt = 0; dt < 4; ++dt) {
      const bf16_t* vr = V + (size_t)(dt * 16 + ln) * SEQ + kb + g * 16;
      v16bf bV = *(const v16bf*)vr;
      ctx[dt] = __builtin_amdgcn_wmma_f32_16x16x32_bf16(
          false, aP, false, bV, (short)0, ctx[dt], false, false);
    }
  }

  // ---- epilogue: normalize and write [B,S,C] (heads already interleaved)
  #pragma unroll
  for (int i = 0; i < 8; ++i) {
    const float inv = 1.0f / lsum[i];
    const int s = q0 + i + 8 * g;
    float* orow = out + ((size_t)b * SEQ + s) * CDIM + h * 64;
    #pragma unroll
    for (int dt = 0; dt < 4; ++dt)
      orow[dt * 16 + ln] = ctx[dt][i] * inv;
  }
}

// ============================================================================
extern "C" void kernel_launch(void* const* d_in, const int* in_sizes, int n_in,
                              void* d_out, int out_size, void* d_ws, size_t ws_size,
                              hipStream_t stream) {
  const float* hidden = (const float*)d_in[0];  // [4,1024,1024]
  const float* amask  = (const float*)d_in[1];  // [4,1,1,1024]
  const float* rel    = (const float*)d_in[2];  // [4,16,1024,1024]
  const float* rel2   = (const float*)d_in[3];  // [4,16,1024,1024]
  const float* Wq     = (const float*)d_in[4];
  const float* bq     = (const float*)d_in[5];
  const float* Wk     = (const float*)d_in[6];
  const float* bk     = (const float*)d_in[7];
  const float* Wv     = (const float*)d_in[8];
  const float* bv     = (const float*)d_in[9];
  float* out = (float*)d_out;

  const size_t nQKV = (size_t)BATCH * HEADS * SEQ * DHEAD;  // 4M elems, 8MB bf16
  bf16_t* Qw = (bf16_t*)d_ws;
  bf16_t* Kw = Qw + nQKV;
  bf16_t* Vw = Kw + nQKV;   // total 24MB of workspace

  // Kernel 1: (256 m-tiles / 4 per block) x 16 heads x {Q,K,V}
  qkv_proj_kernel<<<dim3(64, 16, 3), 128, 0, stream>>>(
      hidden, Wq, bq, Wk, bk, Wv, bv, Qw, Kw, Vw);

  // Kernel 2: 64 (b,h) pairs x (64 query tiles / 4 per block)
  flash_attn_kernel<<<dim3(64, 16), 128, 0, stream>>>(
      Qw, Kw, Vw, rel, rel2, amask, out);
}
